// SpecificMoE_61615600828918
// MI455X (gfx1250) — compile-verified, matrix operand order checked by reading, not run on hardware
//
#include <hip/hip_runtime.h>
#include <hip/hip_bf16.h>
#include <math.h>

#define NTOK   32768          // B*S = 8*4096
#define DDIM   512
#define HDIM   2048
#define NEXP   6
#define TM     128            // tokens per block tile
#define NTHREADS 512          // 16 waves
#define MAX_TILES ((2 * NTOK) / TM + NEXP)
// dynamic LDS: x tile + h chunk + token idx + token weight
#define SMEM_BYTES ((size_t)TM * DDIM * 2 + (size_t)TM * 64 * 2 + TM * 4 + TM * 4)

typedef __attribute__((ext_vector_type(16))) __bf16 v16bf;
typedef __attribute__((ext_vector_type(8)))  float  v8f;

__device__ __forceinline__ v8f wmma_bf16(v16bf a, v16bf b, v8f c) {
  // (neg_a, A, neg_b, B, c_mod, C, reuse_a, reuse_b)
  return __builtin_amdgcn_wmma_f32_16x16x32_bf16(false, a, false, b, (short)0, c, false, false);
}

__device__ __forceinline__ unsigned short f2bf(float f) {
  unsigned int u = __float_as_uint(f);
  u += 0x7FFFu + ((u >> 16) & 1u);   // round to nearest even
  return (unsigned short)(u >> 16);
}

// ------------------------------------------------ plain f32 -> bf16 (for x)
__global__ void cvt_bf16_kernel(const float* __restrict__ src,
                                unsigned short* __restrict__ dst, int n) {
  for (int i = blockIdx.x * blockDim.x + threadIdx.x; i < n;
       i += gridDim.x * blockDim.x)
    dst[i] = f2bf(src[i]);
}

// ------------------------------------------------ weights -> bf16 WMMA-fragment order
// dst chunk o (16 bf16) = B-fragment slice for (e, row-tile rt, k-chunk kc, lane):
//   element (K = kc*32 + (lane>>4)*16 + j, N-row = rt*16 + (lane&15)) of W[e] (R x C, row-major)
__global__ void cvt_wfrag_kernel(const float* __restrict__ src,
                                 unsigned short* __restrict__ dst,
                                 int R, int C) {
  int nchunks = NEXP * (R >> 4) * (C >> 5) * 32;
  int ckc = C >> 5, crt = R >> 4;
  for (int o = blockIdx.x * blockDim.x + threadIdx.x; o < nchunks;
       o += gridDim.x * blockDim.x) {
    int lane = o & 31;
    int kc   = (o >> 5) % ckc;
    int rt   = ((o >> 5) / ckc) % crt;
    int e    = (o >> 5) / (ckc * crt);
    const float* s = src + ((size_t)e * R + rt * 16 + (lane & 15)) * C
                         + kc * 32 + (lane >> 4) * 16;
    unsigned short* d = dst + (size_t)o * 16;
#pragma unroll
    for (int j = 0; j < 16; j++) d[j] = f2bf(s[j]);
  }
}

// ------------------------------------------------ router
__global__ void router_kernel(const float* __restrict__ x,
                              const float* __restrict__ rw,
                              float* __restrict__ probs,
                              int* __restrict__ tok_idx,
                              float* __restrict__ tok_w,
                              int* __restrict__ counts) {
  __shared__ float rwl[NEXP * DDIM];                 // 12 KB
  int tid = threadIdx.x;
  for (int i = tid; i < NEXP * DDIM; i += blockDim.x) rwl[i] = rw[i];
  __syncthreads();

  int wave = tid >> 5, lane = tid & 31;
  int wtot = gridDim.x * (blockDim.x >> 5);
  for (int n = blockIdx.x * (blockDim.x >> 5) + wave; n < NTOK; n += wtot) {
    const float4* xr = (const float4*)(x + (size_t)n * DDIM + lane * 16);
    float acc[NEXP];
#pragma unroll
    for (int e = 0; e < NEXP; e++) acc[e] = 0.f;
#pragma unroll
    for (int q = 0; q < 4; q++) {
      float4 xv = xr[q];
#pragma unroll
      for (int e = 0; e < NEXP; e++) {
        const float* r = &rwl[e * DDIM + lane * 16 + q * 4];
        acc[e] += xv.x * r[0] + xv.y * r[1] + xv.z * r[2] + xv.w * r[3];
      }
    }
#pragma unroll
    for (int off = 16; off > 0; off >>= 1)
#pragma unroll
      for (int e = 0; e < NEXP; e++) acc[e] += __shfl_xor(acc[e], off, 32);

    if (lane == 0) {
      float mx = acc[0];
      for (int e = 1; e < NEXP; e++) mx = fmaxf(mx, acc[e]);
      float p[NEXP], s = 0.f;
      for (int e = 0; e < NEXP; e++) { p[e] = expf(acc[e] - mx); s += p[e]; }
      float inv = 1.f / s;
      for (int e = 0; e < NEXP; e++) {
        p[e] *= inv;
        probs[(size_t)n * NEXP + e] = p[e];
      }
      int i0 = 0;
      for (int e = 1; e < NEXP; e++) if (p[e] > p[i0]) i0 = e;
      int i1 = (i0 == 0) ? 1 : 0;
      for (int e = 0; e < NEXP; e++) if (e != i0 && p[e] > p[i1]) i1 = e;
      float rinv = 1.f / (p[i0] + p[i1]);
      int pos0 = atomicAdd(&counts[i0], 1);
      tok_idx[i0 * NTOK + pos0] = n;  tok_w[i0 * NTOK + pos0] = p[i0] * rinv;
      int pos1 = atomicAdd(&counts[i1], 1);
      tok_idx[i1 * NTOK + pos1] = n;  tok_w[i1 * NTOK + pos1] = p[i1] * rinv;
    }
  }
}

// ------------------------------------------------ tile table
__global__ void build_tiles_kernel(const int* __restrict__ counts,
                                   int2* __restrict__ meta,
                                   int* __restrict__ total) {
  if (threadIdx.x == 0 && blockIdx.x == 0) {
    int t = 0;
    for (int e = 0; e < NEXP; e++) {
      int c = counts[e]; if (c > NTOK) c = NTOK;
      int nt = (c + TM - 1) / TM;
      for (int i = 0; i < nt; i++) { meta[t].x = e; meta[t].y = i * TM; t++; }
    }
    *total = t;
  }
}

// ------------------------------------------------ fused expert kernel
// Block: 128 tokens x full D=512, 16 waves.
// Per 64-wide H chunk:
//   GEMM1: wave owns (ht = wave&3, row-tile pair rt0 = (wave>>2)*2) -> shares one B fragment
//   GEMM2: wave owns 2 col-tiles (wave*32..+32) x 8 row-tiles, B fragments coalesced
__global__ __launch_bounds__(NTHREADS) void moe_expert_kernel(
    const unsigned short* __restrict__ xbf,
    const unsigned short* __restrict__ w1f,
    const unsigned short* __restrict__ w2f,
    const float* __restrict__ b1, const float* __restrict__ b2,
    const int* __restrict__ tok_idx, const float* __restrict__ tok_w,
    const int* __restrict__ counts, const int2* __restrict__ meta,
    const int* __restrict__ total, float* __restrict__ out) {
  extern __shared__ char smem[];
  unsigned short* xs  = (unsigned short*)smem;                              // 128 KB
  unsigned short* hsm = (unsigned short*)(smem + (size_t)TM * DDIM * 2);    //  16 KB
  int*   tidx = (int*)(smem + (size_t)TM * DDIM * 2 + (size_t)TM * 64 * 2);
  float* twt  = (float*)((char*)tidx + TM * 4);

  if (blockIdx.x >= *total) return;
  int2 m = meta[blockIdx.x];
  int e = m.x, start = m.y;
  int cnt = counts[e]; if (cnt > NTOK) cnt = NTOK;

  int tid = threadIdx.x, wave = tid >> 5, lane = tid & 31;
  int ncol = lane & 15, khalf = lane >> 4;

  if (tid < TM) {
    int i = start + tid;
    if (i < cnt) { tidx[tid] = tok_idx[e * NTOK + i]; twt[tid] = tok_w[e * NTOK + i]; }
    else         { tidx[tid] = -1;                    twt[tid] = 0.f; }
  }
  __syncthreads();

  { // stage x tile: 128 rows x 1 KB, 4 threads per row, 256 B each
    int row = tid >> 2, seg = tid & 3;
    int t = tidx[row];
    uint4* dst = (uint4*)&xs[row * DDIM + seg * 128];
    if (t >= 0) {
      const uint4* src = (const uint4*)&xbf[(size_t)t * DDIM + seg * 128];
#pragma unroll
      for (int j = 0; j < 16; j++) dst[j] = src[j];
    } else {
      uint4 z = {0, 0, 0, 0};
#pragma unroll
      for (int j = 0; j < 16; j++) dst[j] = z;
    }
  }
  __syncthreads();

  const unsigned short* w1e = w1f + (size_t)e * (HDIM / 16) * (DDIM / 32) * 512;
  const unsigned short* w2e = w2f + (size_t)e * (DDIM / 16) * (HDIM / 32) * 512;

  v8f acc2[8][2];
#pragma unroll
  for (int a = 0; a < 8; a++)
#pragma unroll
    for (int b = 0; b < 2; b++) acc2[a][b] = (v8f){0, 0, 0, 0, 0, 0, 0, 0};

  int ht  = wave & 3;              // GEMM1 hh col-tile within chunk
  int rt0 = (wave >> 2) * 2;       // GEMM1 row-tile pair

  for (int hh0 = 0; hh0 < HDIM; hh0 += 64) {
    if (hh0 + 64 < HDIM) {  // prefetch next chunk's fragments (global_prefetch_b8)
      __builtin_prefetch(&w1e[((size_t)((hh0 >> 4) + 4 + ht) * (DDIM / 32)) * 512 + lane * 16], 0, 1);
      __builtin_prefetch(&w2e[((size_t)(wave * 2) * (HDIM / 32) + (hh0 >> 5) + 2) * 512 + lane * 16], 0, 1);
    }

    // ---- GEMM1: two 16x16 h tiles sharing one B fragment stream
    v8f acc1a = (v8f){0, 0, 0, 0, 0, 0, 0, 0};
    v8f acc1b = (v8f){0, 0, 0, 0, 0, 0, 0, 0};
    const unsigned short* bfrag = w1e + ((size_t)((hh0 >> 4) + ht) * (DDIM / 32)) * 512 + lane * 16;
    const unsigned short* xspa = &xs[(rt0 * 16 + ncol) * DDIM + khalf * 16];
    const unsigned short* xspb = xspa + 16 * DDIM;
#pragma unroll 4
    for (int kc = 0; kc < DDIM / 32; kc++) {
      v16bf b  = *(const v16bf*)(bfrag + (size_t)kc * 512);   // coalesced burst
      v16bf aa = *(const v16bf*)(xspa + kc * 32);
      v16bf ab = *(const v16bf*)(xspb + kc * 32);
      acc1a = wmma_bf16(aa, b, acc1a);
      acc1b = wmma_bf16(ab, b, acc1b);
    }
    float b1v = b1[e * HDIM + hh0 + ht * 16 + ncol];
    int hcol = ht * 16 + ncol;
#pragma unroll
    for (int r = 0; r < 8; r++) {
      float va = acc1a[r] + b1v;
      va = 0.5f * va * (1.f + erff(va * 0.70710678118f));
      hsm[(rt0 * 16 + khalf * 8 + r) * 64 + hcol] = f2bf(va);
      float vb = acc1b[r] + b1v;
      vb = 0.5f * vb * (1.f + erff(vb * 0.70710678118f));
      hsm[((rt0 + 1) * 16 + khalf * 8 + r) * 64 + hcol] = f2bf(vb);
    }
    __syncthreads();

    // ---- GEMM2: 2 col-tiles x 8 row-tiles, K over this 64-wide H chunk
#pragma unroll
    for (int kk = 0; kk < 2; kk++) {
      v16bf bf0 = *(const v16bf*)&w2e[((size_t)(wave * 2 + 0) * (HDIM / 32) + (hh0 >> 5) + kk) * 512 + lane * 16];
      v16bf bf1 = *(const v16bf*)&w2e[((size_t)(wave * 2 + 1) * (HDIM / 32) + (hh0 >> 5) + kk) * 512 + lane * 16];
#pragma unroll
      for (int rt = 0; rt < 8; rt++) {
        v16bf a = *(const v16bf*)&hsm[(rt * 16 + ncol) * 64 + kk * 32 + khalf * 16];
        acc2[rt][0] = wmma_bf16(a, bf0, acc2[rt][0]);
        acc2[rt][1] = wmma_bf16(a, bf1, acc2[rt][1]);
      }
    }
    __syncthreads();
  }

  // epilogue: out[token] += w * (acc + b2)
  float b2v0 = b2[e * DDIM + wave * 32 + ncol];
  float b2v1 = b2[e * DDIM + wave * 32 + 16 + ncol];
#pragma unroll
  for (int rt = 0; rt < 8; rt++) {
#pragma unroll
    for (int r = 0; r < 8; r++) {
      int row = rt * 16 + khalf * 8 + r;
      int t = tidx[row];
      if (t < 0) continue;
      float w = twt[row];
      atomicAdd(&out[(size_t)t * DDIM + wave * 32 + ncol],      w * (acc2[rt][0][r] + b2v0));
      atomicAdd(&out[(size_t)t * DDIM + wave * 32 + 16 + ncol], w * (acc2[rt][1][r] + b2v1));
    }
  }
}

// ------------------------------------------------ launcher
extern "C" void kernel_launch(void* const* d_in, const int* in_sizes, int n_in,
                              void* d_out, int out_size, void* d_ws, size_t ws_size,
                              hipStream_t stream) {
  (void)in_sizes; (void)n_in; (void)out_size; (void)ws_size;
  const float* x  = (const float*)d_in[0];
  const float* rw = (const float*)d_in[1];
  const float* w1 = (const float*)d_in[2];
  const float* b1 = (const float*)d_in[3];
  const float* w2 = (const float*)d_in[4];
  const float* b2 = (const float*)d_in[5];

  float* out   = (float*)d_out;                   // [NTOK, DDIM]
  float* probs = out + (size_t)NTOK * DDIM;       // [NTOK, NEXP]

  char* ws = (char*)d_ws;
  auto align = [](size_t v) { return (v + 255) & ~(size_t)255; };
  size_t off = 0;
  unsigned short* xbf = (unsigned short*)(ws + off); off = align(off + (size_t)NTOK * DDIM * 2);
  unsigned short* w1f = (unsigned short*)(ws + off); off = align(off + (size_t)NEXP * HDIM * DDIM * 2);
  unsigned short* w2f = (unsigned short*)(ws + off); off = align(off + (size_t)NEXP * DDIM * HDIM * 2);
  int*   tok_idx = (int*)(ws + off);   off = align(off + (size_t)NEXP * NTOK * 4);
  float* tok_w   = (float*)(ws + off); off = align(off + (size_t)NEXP * NTOK * 4);
  int*   counts  = (int*)(ws + off);   off = align(off + 256);
  int*   total   = (int*)(ws + off);   off = align(off + 256);
  int2*  meta    = (int2*)(ws + off);  off = align(off + (size_t)MAX_TILES * sizeof(int2));

  hipMemsetAsync(out, 0, (size_t)NTOK * DDIM * sizeof(float), stream);
  hipMemsetAsync(counts, 0, 512, stream);   // counts + total

  cvt_bf16_kernel<<<2048, 256, 0, stream>>>(x, xbf, NTOK * DDIM);
  cvt_wfrag_kernel<<<1024, 256, 0, stream>>>(w1, w1f, HDIM, DDIM);  // B frags for GEMM1
  cvt_wfrag_kernel<<<1024, 256, 0, stream>>>(w2, w2f, DDIM, HDIM);  // B frags for GEMM2

  router_kernel<<<512, 256, 0, stream>>>(x, rw, probs, tok_idx, tok_w, counts);
  build_tiles_kernel<<<1, 64, 0, stream>>>(counts, meta, total);
  moe_expert_kernel<<<MAX_TILES, NTHREADS, SMEM_BYTES, stream>>>(
      xbf, w1f, w2f, b1, b2, tok_idx, tok_w, counts, meta, total, out);
}
